// PLE_7705171329625
// MI455X (gfx1250) — matrix operand order repaired and checked by reading, not compiled, not run on hardware
//
#include <hip/hip_runtime.h>
#include <hip/hip_bf16.h>

// ---------------------------------------------------------------------------
// PLE (Progressive Layered Extraction), 2 layers.
// B=4096, D=H=1024, T=4, NTE=2, NSE=2 -> E=10 experts, K=4 selected per task.
// EXPERT_IN_IDX[e] = e>>1 ; SEL[t] = {2t, 2t+1, 8, 9}
// Dominant cost: 2 layers x 10 experts x (4096x1024x1024) GEMM = 172 GFLOP.
// Split-bf16 WMMA (Ah*Bh + Al*Bh + Ah*Bl, f32 accum) => near-f32 accuracy at
// bf16 tensor rates. B tiles staged via GLOBAL_LOAD_ASYNC_TO_LDS (ASYNCcnt).
// ---------------------------------------------------------------------------

typedef __attribute__((ext_vector_type(16))) __bf16 v16bf;
typedef __attribute__((ext_vector_type(8)))  __bf16 v8bf;
typedef __attribute__((ext_vector_type(8)))  float  v8f;
typedef __attribute__((ext_vector_type(4)))  float  v4f;

__device__ inline v16bf cat16(v8bf a, v8bf b) {
  return __builtin_shufflevector(a, b, 0,1,2,3,4,5,6,7,8,9,10,11,12,13,14,15);
}

// Async DMA: 16B per lane, global -> LDS, tracked by ASYNCcnt.
__device__ inline void async_copy_b128(unsigned lds_off, const void* gaddr) {
  asm volatile("global_load_async_to_lds_b128 %0, %1, off"
               :: "v"(lds_off), "v"((unsigned long long)(uintptr_t)gaddr)
               : "memory");
}
__device__ inline void wait_async0() {
  asm volatile("s_wait_asynccnt 0x0" ::: "memory");
}

// ---------------------------------------------------------------------------
// Transpose (D x H) -> (H x D) per expert and split f32 -> bf16 hi/lo pair.
// grid (H/32, D/32, E), block (32,8).
// ---------------------------------------------------------------------------
__global__ __launch_bounds__(256)
void transpose_split(const float* __restrict__ W,
                     __bf16* __restrict__ Th, __bf16* __restrict__ Tl) {
  __shared__ float tile[32][33];
  const int e  = blockIdx.z;
  const int n0 = blockIdx.x * 32;   // output-row (h) block
  const int d0 = blockIdx.y * 32;   // k block
  const int tx = threadIdx.x, ty = threadIdx.y;
  const float* src = W + (size_t)e * 1024 * 1024;
#pragma unroll
  for (int i = 0; i < 4; ++i) {
    int dd = ty + i * 8;
    tile[dd][tx] = src[(size_t)(d0 + dd) * 1024 + n0 + tx];
  }
  __syncthreads();
#pragma unroll
  for (int i = 0; i < 4; ++i) {
    int nn = ty + i * 8;
    float v  = tile[tx][nn];          // = src[d0+tx][n0+nn]
    __bf16 h = (__bf16)v;
    __bf16 l = (__bf16)(v - (float)h);
    size_t o = (size_t)e * 1024 * 1024 + (size_t)(n0 + nn) * 1024 + d0 + tx;
    Th[o] = h;
    Tl[o] = l;
  }
}

// ---------------------------------------------------------------------------
// Expert GEMM: eo[b,e,:] = relu( X[b*5 + (e>>1), :] @ Bt[e]^T + bias[e,:] )
// X:  f32, logical row = b*5 + (e>>1)
// Bt: bf16 hi/lo, (E, H=1024 rows, D=1024), row-major (pre-transposed W)
// eo: f32, (B, E=10, H)
// grid (H/128, B/128, E), block 256 (8 waves); wave tile 32x64 (2x4 WMMA).
// ---------------------------------------------------------------------------
__global__ __launch_bounds__(256)
void ple_gemm(const float* __restrict__ X,
              const __bf16* __restrict__ Bth, const __bf16* __restrict__ Btl,
              const float* __restrict__ bias, float* __restrict__ eo) {
  constexpr int LDT = 40;                       // padded row (32 data + 8), 80B
  __shared__ __bf16 sAh[128 * LDT];
  __shared__ __bf16 sAl[128 * LDT];
  __shared__ __bf16 sBh[128 * LDT];
  __shared__ __bf16 sBl[128 * LDT];

  const int tid  = threadIdx.x;
  const int lane = tid & 31;
  const int wave = tid >> 5;
  const int wm   = wave & 3;                    // 32-row slice within 128
  const int wn   = wave >> 2;                   // 64-col slice within 128
  const int e    = blockIdx.z;
  const int m0   = blockIdx.y * 128;
  const int n0   = blockIdx.x * 128;
  const int eidx = e >> 1;                      // EXPERT_IN_IDX

  // cooperative tile loaders: 256 threads = 128 rows x 2 sixteen-wide segments
  const int lrow = tid >> 1;
  const int lseg = (tid & 1) * 16;
  const float*  aPtr = X   + ((size_t)(m0 + lrow) * 5 + eidx) * 1024 + lseg;
  const __bf16* bhP  = Bth + ((size_t)e * 1024 + n0 + lrow) * 1024 + lseg;
  const __bf16* blP  = Btl + ((size_t)e * 1024 + n0 + lrow) * 1024 + lseg;
  const int sOff = lrow * LDT + lseg;
  const unsigned sBhOff = (unsigned)(uintptr_t)(sBh + sOff);
  const unsigned sBlOff = (unsigned)(uintptr_t)(sBl + sOff);

  v8f acc[2][4];
#pragma unroll
  for (int m = 0; m < 2; ++m)
#pragma unroll
    for (int n = 0; n < 4; ++n)
#pragma unroll
      for (int i = 0; i < 8; ++i) acc[m][n][i] = 0.f;

  const int frow = lane & 15;
  const int half = lane >> 4;                   // 0: lanes 0-15, 1: lanes 16-31

  for (int ks = 0; ks < 32; ++ks) {
    const int k0 = ks * 32;
    // ---- A tile: f32 global load, split into bf16 hi/lo in registers ----
    v4f a0 = *(const v4f*)(aPtr + k0);
    v4f a1 = *(const v4f*)(aPtr + k0 + 4);
    v4f a2 = *(const v4f*)(aPtr + k0 + 8);
    v4f a3 = *(const v4f*)(aPtr + k0 + 12);
    if (ks + 1 < 32) {
      __builtin_prefetch(aPtr + k0 + 32, 0, 3);   // -> global_prefetch_b8
      __builtin_prefetch(bhP  + k0 + 32, 0, 3);
      __builtin_prefetch(blP  + k0 + 32, 0, 3);
    }
    v8bf ah0, ah1, al0, al1;
#pragma unroll
    for (int j = 0; j < 4; ++j) {
      float v; __bf16 hh;
      v = a0[j]; hh = (__bf16)v; ah0[j]     = hh; al0[j]     = (__bf16)(v - (float)hh);
      v = a1[j]; hh = (__bf16)v; ah0[j + 4] = hh; al0[j + 4] = (__bf16)(v - (float)hh);
      v = a2[j]; hh = (__bf16)v; ah1[j]     = hh; al1[j]     = (__bf16)(v - (float)hh);
      v = a3[j]; hh = (__bf16)v; ah1[j + 4] = hh; al1[j + 4] = (__bf16)(v - (float)hh);
    }

    __syncthreads();                            // prev tiles fully consumed

    // ---- B tiles: async DMA global -> LDS (no VGPR staging, ASYNCcnt) ----
    async_copy_b128(sBhOff,      bhP + k0);
    async_copy_b128(sBhOff + 16, bhP + k0 + 8);
    async_copy_b128(sBlOff,      blP + k0);
    async_copy_b128(sBlOff + 16, blP + k0 + 8);

    // ---- A tiles: ds_store of converted hi/lo ----
    *(v8bf*)(sAh + sOff) = ah0;  *(v8bf*)(sAh + sOff + 8) = ah1;
    *(v8bf*)(sAl + sOff) = al0;  *(v8bf*)(sAl + sOff + 8) = al1;

    wait_async0();                              // own async copies landed
    __syncthreads();                            // everyone's tiles visible

    // ---- fragment loads ----
    // A (16-bit A 16x32): lanes 0-15 hold K{0..7,16..23}, lanes 16-31 hold
    // K{8..15,24..31}; two 16B LDS reads at +8*half and +8*half+16.
    v16bf ah[2], al[2];
#pragma unroll
    for (int m = 0; m < 2; ++m) {
      int r   = wm * 32 + m * 16 + frow;
      int off = r * LDT + half * 8;
      ah[m] = cat16(*(const v8bf*)(sAh + off), *(const v8bf*)(sAh + off + 16));
      al[m] = cat16(*(const v8bf*)(sAl + off), *(const v8bf*)(sAl + off + 16));
    }
    // B (16-bit B 32x16): lanes 0-15 hold K{0..7,16..23}, lanes 16-31 hold
    // K{8..15,24..31}; two 16B LDS reads at +8*half and +8*half+16.
    v16bf bh[4], bl[4];
#pragma unroll
    for (int n = 0; n < 4; ++n) {
      int r   = wn * 64 + n * 16 + frow;
      int off = r * LDT + half * 8;
      bh[n] = cat16(*(const v8bf*)(sBh + off), *(const v8bf*)(sBh + off + 16));
      bl[n] = cat16(*(const v8bf*)(sBl + off), *(const v8bf*)(sBl + off + 16));
    }

    // ---- split-bf16 WMMA: acc += Ah*Bh + Al*Bh + Ah*Bl ----
#pragma unroll
    for (int m = 0; m < 2; ++m)
#pragma unroll
      for (int n = 0; n < 4; ++n) {
        acc[m][n] = __builtin_amdgcn_wmma_f32_16x16x32_bf16(
            false, ah[m], false, bh[n], (short)0, acc[m][n], false, false);
        acc[m][n] = __builtin_amdgcn_wmma_f32_16x16x32_bf16(
            false, al[m], false, bh[n], (short)0, acc[m][n], false, false);
        acc[m][n] = __builtin_amdgcn_wmma_f32_16x16x32_bf16(
            false, ah[m], false, bl[n], (short)0, acc[m][n], false, false);
      }
  }

  // ---- epilogue: bias + ReLU, C/D layout: lane n=lane&15, vgpr r -> M=r+8*half
#pragma unroll
  for (int n = 0; n < 4; ++n) {
    int col  = n0 + wn * 64 + n * 16 + (lane & 15);
    float bv = bias[e * 1024 + col];
#pragma unroll
    for (int m = 0; m < 2; ++m) {
      int rbase = m0 + wm * 32 + m * 16 + half * 8;
#pragma unroll
      for (int r = 0; r < 8; ++r) {
        float v = acc[m][n][r] + bv;
        v = v > 0.f ? v : 0.f;
        eo[((size_t)(rbase + r) * 10 + e) * 1024 + col] = v;
      }
    }
  }
}

// ---------------------------------------------------------------------------
// Task gates: g[b,t,k] = softmax_k( X[b*5+t,:] . Gw[t,:,k] + Gb[t,k] )
// ---------------------------------------------------------------------------
__global__ __launch_bounds__(256)
void gate_task(const float* __restrict__ X, const float* __restrict__ Gw,
               const float* __restrict__ Gb, float* __restrict__ G) {
  int idx = blockIdx.x * 256 + threadIdx.x;           // B*T
  if (idx >= 4096 * 4) return;
  int b = idx >> 2, t = idx & 3;
  const float* xr = X  + ((size_t)b * 5 + t) * 1024;
  const float* wr = Gw + (size_t)t * 1024 * 4;
  float a0 = 0.f, a1 = 0.f, a2 = 0.f, a3 = 0.f;
  for (int d = 0; d < 1024; ++d) {
    float xv = xr[d];
    v4f w = *(const v4f*)(wr + d * 4);
    a0 += xv * w[0]; a1 += xv * w[1]; a2 += xv * w[2]; a3 += xv * w[3];
  }
  a0 += Gb[t * 4 + 0]; a1 += Gb[t * 4 + 1];
  a2 += Gb[t * 4 + 2]; a3 += Gb[t * 4 + 3];
  float mx = fmaxf(fmaxf(a0, a1), fmaxf(a2, a3));
  float e0 = expf(a0 - mx), e1 = expf(a1 - mx);
  float e2 = expf(a2 - mx), e3 = expf(a3 - mx);
  float inv = 1.f / (e0 + e1 + e2 + e3);
  G[(size_t)idx * 4 + 0] = e0 * inv; G[(size_t)idx * 4 + 1] = e1 * inv;
  G[(size_t)idx * 4 + 2] = e2 * inv; G[(size_t)idx * 4 + 3] = e3 * inv;
}

// ---------------------------------------------------------------------------
// Shared gate (layer 0 only): gs[b,e] = softmax_e( X[b*5+4,:] @ Gsw + Gsb )
// ---------------------------------------------------------------------------
__global__ __launch_bounds__(256)
void gate_shared(const float* __restrict__ X, const float* __restrict__ Gsw,
                 const float* __restrict__ Gsb, float* __restrict__ Gs) {
  int b = blockIdx.x * 256 + threadIdx.x;
  if (b >= 4096) return;
  const float* xr = X + ((size_t)b * 5 + 4) * 1024;
  float acc[10];
#pragma unroll
  for (int e = 0; e < 10; ++e) acc[e] = 0.f;
  for (int d = 0; d < 1024; ++d) {
    float xv = xr[d];
    const float* wr = Gsw + (size_t)d * 10;
#pragma unroll
    for (int e = 0; e < 10; ++e) acc[e] += xv * wr[e];
  }
  float mx = -3.0e38f;
#pragma unroll
  for (int e = 0; e < 10; ++e) { acc[e] += Gsb[e]; mx = fmaxf(mx, acc[e]); }
  float s = 0.f;
#pragma unroll
  for (int e = 0; e < 10; ++e) { acc[e] = expf(acc[e] - mx); s += acc[e]; }
  float inv = 1.f / s;
#pragma unroll
  for (int e = 0; e < 10; ++e) Gs[(size_t)b * 10 + e] = acc[e] * inv;
}

// ---------------------------------------------------------------------------
// Layer-0 combine: h[b,t,:] = sum_k g[b,t,k]*eo[b,SEL[t,k],:]  (t<4)
//                  h[b,4,:] = sum_e gs[b,e]*eo[b,e,:]
// ---------------------------------------------------------------------------
__global__ __launch_bounds__(256)
void combine0(const float* __restrict__ eo, const float* __restrict__ g,
              const float* __restrict__ gs, float* __restrict__ h) {
  int row = blockIdx.x;                  // b*5 + t
  int b = row / 5, t = row - b * 5;
  int col = threadIdx.x;
  if (t < 4) {
    const float* gp = g + ((size_t)b * 4 + t) * 4;
    float s0 = gp[0], s1 = gp[1], s2 = gp[2], s3 = gp[3];
    const float* p0 = eo + ((size_t)b * 10 + 2 * t    ) * 1024;
    const float* p1 = eo + ((size_t)b * 10 + 2 * t + 1) * 1024;
    const float* p2 = eo + ((size_t)b * 10 + 8) * 1024;
    const float* p3 = eo + ((size_t)b * 10 + 9) * 1024;
#pragma unroll
    for (int j = 0; j < 4; ++j) {
      int c = col + j * 256;
      h[(size_t)row * 1024 + c] = s0 * p0[c] + s1 * p1[c] + s2 * p2[c] + s3 * p3[c];
    }
  } else {
    const float* gp = gs + (size_t)b * 10;
#pragma unroll
    for (int j = 0; j < 4; ++j) {
      int c = col + j * 256;
      float acc = 0.f;
#pragma unroll
      for (int e = 0; e < 10; ++e)
        acc += gp[e] * eo[((size_t)b * 10 + e) * 1024 + c];
      h[(size_t)row * 1024 + c] = acc;
    }
  }
}

// ---------------------------------------------------------------------------
// Layer-1 combine into final output (B,T,H).
// ---------------------------------------------------------------------------
__global__ __launch_bounds__(256)
void combine1(const float* __restrict__ eo, const float* __restrict__ g,
              float* __restrict__ out) {
  int idx = blockIdx.x;                  // b*4 + t
  int b = idx >> 2, t = idx & 3;
  int col = threadIdx.x;
  const float* gp = g + (size_t)idx * 4;
  float s0 = gp[0], s1 = gp[1], s2 = gp[2], s3 = gp[3];
  const float* p0 = eo + ((size_t)b * 10 + 2 * t    ) * 1024;
  const float* p1 = eo + ((size_t)b * 10 + 2 * t + 1) * 1024;
  const float* p2 = eo + ((size_t)b * 10 + 8) * 1024;
  const float* p3 = eo + ((size_t)b * 10 + 9) * 1024;
#pragma unroll
  for (int j = 0; j < 4; ++j) {
    int c = col + j * 256;
    out[(size_t)idx * 1024 + c] = s0 * p0[c] + s1 * p1[c] + s2 * p2[c] + s3 * p3[c];
  }
}

// ---------------------------------------------------------------------------
extern "C" void kernel_launch(void* const* d_in, const int* in_sizes, int n_in,
                              void* d_out, int out_size, void* d_ws, size_t ws_size,
                              hipStream_t stream) {
  const float* x    = (const float*)d_in[0];   // (B,5,D)
  const float* W0   = (const float*)d_in[1];   // (E,D,H)
  const float* b0   = (const float*)d_in[2];   // (E,H)
  const float* G0w  = (const float*)d_in[3];   // (T,D,K)
  const float* G0b  = (const float*)d_in[4];   // (T,K)
  const float* Gs0w = (const float*)d_in[5];   // (D,E)
  const float* Gs0b = (const float*)d_in[6];   // (E)
  const float* W1   = (const float*)d_in[7];   // (E,H,H)
  const float* b1   = (const float*)d_in[8];   // (E,H)
  const float* G1w  = (const float*)d_in[9];   // (T,H,K)
  const float* G1b  = (const float*)d_in[10];  // (T,K)
  float* out = (float*)d_out;

  char* ws = (char*)d_ws;
  size_t off = 0;
  auto alloc = [&](size_t bytes) -> void* {
    void* p = ws + off;
    off += (bytes + 255) & ~(size_t)255;
    return p;
  };
  const size_t WELEM = 10ull * 1024 * 1024;
  __bf16* W0th = (__bf16*)alloc(WELEM * 2);
  __bf16* W0tl = (__bf16*)alloc(WELEM * 2);
  __bf16* W1th = (__bf16*)alloc(WELEM * 2);
  __bf16* W1tl = (__bf16*)alloc(WELEM * 2);
  float*  eo   = (float*)alloc(4096ull * 10 * 1024 * 4);  // reused both layers
  float*  h    = (float*)alloc(4096ull * 5 * 1024 * 4);
  float*  g0   = (float*)alloc(4096ull * 4 * 4 * 4);
  float*  gsv  = (float*)alloc(4096ull * 10 * 4);
  float*  g1   = (float*)alloc(4096ull * 4 * 4 * 4);
  (void)in_sizes; (void)n_in; (void)out_size; (void)ws_size;

  dim3 tsBlk(32, 8);
  dim3 tsGrd(32, 32, 10);
  transpose_split<<<tsGrd, tsBlk, 0, stream>>>(W0, W0th, W0tl);
  transpose_split<<<tsGrd, tsBlk, 0, stream>>>(W1, W1th, W1tl);

  gate_task<<<64, 256, 0, stream>>>(x, G0w, G0b, g0);
  gate_shared<<<16, 256, 0, stream>>>(x, Gs0w, Gs0b, gsv);

  dim3 gemmGrd(1024 / 128, 4096 / 128, 10);
  ple_gemm<<<gemmGrd, 256, 0, stream>>>(x, W0th, W0tl, b0, eo);
  combine0<<<4096 * 5, 256, 0, stream>>>(eo, g0, gsv, h);

  gate_task<<<64, 256, 0, stream>>>(h, G1w, G1b, g1);
  ple_gemm<<<gemmGrd, 256, 0, stream>>>(h, W1th, W1tl, b1, eo);
  combine1<<<4096 * 4, 256, 0, stream>>>(eo, g1, out);
}
